// TreeMLPEncoder_71957882077833
// MI455X (gfx1250) — matrix operand order repaired and checked by reading, not compiled, not run on hardware
//
#include <hip/hip_runtime.h>

typedef __attribute__((ext_vector_type(16))) _Float16 v16h;
typedef __attribute__((ext_vector_type(8)))  _Float16 v8h;
typedef __attribute__((ext_vector_type(8)))  float    v8f;
typedef __attribute__((ext_vector_type(4)))  float    v4f;

#define DIM    256
#define DIM2   512
#define DIM3   768
#define NCHARS 20000
#define CB     16          // chars per block
#define LN_EPS 1e-5f

// padded LDS row strides (in elements) to avoid 64-bank conflicts
#define SX  776            // Xin rows  (768 + 8 halves)
#define SH  520            // H rows    (512 + 8 halves)
#define SYF 260            // Yf rows   (256 + 4 floats)

// LDS layout (bytes)
#define OFF_XIN 0
#define OFF_H   (OFF_XIN + 32 * SX * 2)            // 49664
#define OFF_YF  (OFF_H   + 32 * SH * 2)            // +33280
#define OFF_L1  (OFF_YF  + 32 * SYF * 4)           // +33280
#define SMEM_BYTES (OFF_L1 + 32 * DIM * 2)         // +16384 = 132608

// ---------------------------------------------------------------------------
// Weight packing: f32 row-major [K,N] -> f16 WMMA B-fragment order.
// B fragment (32x16 f16, contiguous k order chosen here, not the ISA table order): lane l holds column n = l%16;
// lanes 0-15 hold K=0..15 (2 per VGPR), lanes 16-31 hold K=16..31.
// Storage: P[((nt*KT + kt)*32 + lane)*16 + (v*2+h)]
// ---------------------------------------------------------------------------
__global__ void pack_weights_kernel(const float* __restrict__ W1,
                                    const float* __restrict__ W2,
                                    _Float16* __restrict__ P1,
                                    _Float16* __restrict__ P2) {
  int idx = blockIdx.x * blockDim.x + threadIdx.x;
  const int n1 = DIM3 * DIM2;                    // 393216 elements for W1
  if (idx < n1) {
    int pos  = idx & 15;
    int lane = (idx >> 4) & 31;
    int frag = idx >> 9;
    int kt = frag % 24, nt = frag / 24;
    int v = pos >> 1, h = pos & 1;
    int k = kt * 32 + v * 2 + h + ((lane >> 4) << 4);
    int n = nt * 16 + (lane & 15);
    P1[idx] = (_Float16)W1[k * DIM2 + n];
  } else {
    int j = idx - n1;
    if (j < DIM2 * DIM) {
      int pos  = j & 15;
      int lane = (j >> 4) & 31;
      int frag = j >> 9;
      int kt = frag % 16, nt = frag / 16;
      int v = pos >> 1, h = pos & 1;
      int k = kt * 32 + v * 2 + h + ((lane >> 4) << 4);
      int n = nt * 16 + (lane & 15);
      P2[j] = (_Float16)W2[k * DIM + n];
    }
  }
}

// ---------------------------------------------------------------------------
// Device helpers
// ---------------------------------------------------------------------------
__device__ __forceinline__ float wave_reduce_add(float x) {
#pragma unroll
  for (int m = 16; m >= 1; m >>= 1) x += __shfl_xor(x, m, 32);
  return x;
}

// A fragment (16x32 f16): lane l holds row m=l%16;
// lanes<16: K = kt*32+0..15; lanes>=16: K = kt*32+16..31 (matches packed B).
__device__ __forceinline__ v16h load_a_frag(const _Float16* buf, int stride,
                                            int m, int kbase, int lane) {
  const _Float16* p = buf + m * stride + kbase + ((lane >> 4) << 4);
  v8h lo = *(const v8h*)(p);
  v8h hi = *(const v8h*)(p + 8);
  v16h a;
#pragma unroll
  for (int e = 0; e < 8; ++e) { a[e] = lo[e]; a[e + 8] = hi[e]; }
  return a;
}

__device__ __forceinline__ v16h load_b_frag(const _Float16* pk, int frag, int lane) {
  return *(const v16h*)(pk + (((size_t)frag * 32 + lane) << 4));
}

// one 16x16 output tile: rows [mt*16, mt*16+16) of Abuf times ntile nt of packed W
__device__ __forceinline__ v8f gemm_tile(const _Float16* Abuf, int sA, int mt,
                                         const _Float16* Pk, int nt, int Kt, int lane) {
  v8f acc = {};
  int m = mt * 16 + (lane & 15);
#pragma unroll
  for (int kt = 0; kt < 24; ++kt) {
    if (kt >= Kt) break;
    v16h a = load_a_frag(Abuf, sA, m, kt * 32, lane);
    v16h b = load_b_frag(Pk, nt * Kt + kt, lane);
    acc = __builtin_amdgcn_wmma_f32_16x16x32_f16(false, a, false, b,
                                                 (short)0, acc, false, false);
  }
  return acc;
}

// two 16x16 output tiles (mt=0 and mt=1) sharing each B fragment:
// halves global B-fragment traffic vs. computing the tiles independently.
__device__ __forceinline__ void gemm_tile_pair(const _Float16* Abuf, int sA,
                                               const _Float16* Pk, int nt, int Kt,
                                               int lane, v8f& acc0, v8f& acc1) {
  int m0 = (lane & 15);
  int m1 = 16 + (lane & 15);
#pragma unroll
  for (int kt = 0; kt < 24; ++kt) {
    if (kt >= Kt) break;
    v16h b  = load_b_frag(Pk, nt * Kt + kt, lane);
    v16h a0 = load_a_frag(Abuf, sA, m0, kt * 32, lane);
    v16h a1 = load_a_frag(Abuf, sA, m1, kt * 32, lane);
    acc0 = __builtin_amdgcn_wmma_f32_16x16x32_f16(false, a0, false, b,
                                                  (short)0, acc0, false, false);
    acc1 = __builtin_amdgcn_wmma_f32_16x16x32_f16(false, a1, false, b,
                                                  (short)0, acc1, false, false);
  }
}

// D layout (ISA 7.12.2): lane l -> N = l%16; VGPR v -> M = v + 8*(l>=16)
__device__ __forceinline__ void store_tile_relu_f16(_Float16* Hbuf, int sH, int mt,
                                                    int nt, v8f acc,
                                                    const float* bias, int lane) {
  int n = nt * 16 + (lane & 15);
  float bv = bias[n];
  int mrow = mt * 16 + ((lane >> 4) << 3);
#pragma unroll
  for (int v = 0; v < 8; ++v) {
    float x = acc[v] + bv;
    Hbuf[(mrow + v) * sH + n] = (_Float16)(x > 0.f ? x : 0.f);
  }
}

__device__ __forceinline__ void store_tile_f32(float* Y, int sY, int mt, int nt,
                                               v8f acc, const float* bias, int lane) {
  int n = nt * 16 + (lane & 15);
  float bv = bias[n];
  int mrow = mt * 16 + ((lane >> 4) << 3);
#pragma unroll
  for (int v = 0; v < 8; ++v) Y[(mrow + v) * sY + n] = acc[v] + bv;
}

// LayerNorm of one 256-float row by one wave (8 elems/lane), f16 destination
__device__ __forceinline__ void ln_row_to_f16(const float* src, _Float16* dst,
                                              const float* gamma, const float* beta,
                                              int lane) {
  float vals[8]; float s = 0.f;
#pragma unroll
  for (int p = 0; p < 8; ++p) { vals[p] = src[lane + 32 * p]; s += vals[p]; }
  s = wave_reduce_add(s);
  float mu = s * (1.0f / 256.0f);
  float q = 0.f;
#pragma unroll
  for (int p = 0; p < 8; ++p) { float d = vals[p] - mu; q += d * d; }
  q = wave_reduce_add(q);
  float rstd = rsqrtf(q * (1.0f / 256.0f) + LN_EPS);
#pragma unroll
  for (int p = 0; p < 8; ++p) {
    int k = lane + 32 * p;
    dst[k] = (_Float16)((vals[p] - mu) * rstd * gamma[k] + beta[k]);
  }
}

__device__ __forceinline__ void ln_row_to_f32(const float* src, float* dst,
                                              const float* gamma, const float* beta,
                                              int lane) {
  float vals[8]; float s = 0.f;
#pragma unroll
  for (int p = 0; p < 8; ++p) { vals[p] = src[lane + 32 * p]; s += vals[p]; }
  s = wave_reduce_add(s);
  float mu = s * (1.0f / 256.0f);
  float q = 0.f;
#pragma unroll
  for (int p = 0; p < 8; ++p) { float d = vals[p] - mu; q += d * d; }
  q = wave_reduce_add(q);
  float rstd = rsqrtf(q * (1.0f / 256.0f) + LN_EPS);
#pragma unroll
  for (int p = 0; p < 8; ++p) {
    int k = lane + 32 * p;
    dst[k] = (vals[p] - mu) * rstd * gamma[k] + beta[k];
  }
}

// ---------------------------------------------------------------------------
// Main: per block, evaluate the tree MLP for CB=16 chars -> root_table rows
// ---------------------------------------------------------------------------
__global__ void tree_encode_kernel(const int* __restrict__ leaf_ids,
                                   const int* __restrict__ op_ids,
                                   const float* __restrict__ comp_emb,
                                   const float* __restrict__ op_emb,
                                   const _Float16* __restrict__ P1,
                                   const _Float16* __restrict__ P2,
                                   const float* __restrict__ b1,
                                   const float* __restrict__ b2,
                                   const float* __restrict__ gamma,
                                   const float* __restrict__ beta,
                                   float* __restrict__ table) {
  extern __shared__ char smem[];
  _Float16* Xin = (_Float16*)(smem + OFF_XIN);   // [32][SX]  l1_in / root_in
  _Float16* H   = (_Float16*)(smem + OFF_H);     // [32][SH]  hidden
  float*    Yf  = (float*)(smem + OFF_YF);       // [32][SYF] pre-LN output
  _Float16* L1h = (_Float16*)(smem + OFF_L1);    // [32][256] LN'd l1

  const int lane = threadIdx.x & 31;
  const int wid  = threadIdx.x >> 5;             // 0..7
  const int c0   = blockIdx.x * CB;

  // ---- Stage 0: build l1_in rows [32][768]
  // 64 leaf rows (LN'd comp_emb), 8 per wave
#pragma unroll 1
  for (int i = 0; i < 8; ++i) {
    int r = wid * 8 + i;                         // 0..63
    int c = r >> 2, j = r & 3;
    int lid = leaf_ids[(c0 + c) * 4 + j];
    const float* src = comp_emb + (size_t)lid * DIM;
    _Float16* dst = Xin + (2 * c + (j >> 1)) * SX + DIM + (j & 1) * DIM;
    ln_row_to_f16(src, dst, gamma, beta, lane);
  }
  // 32 op rows (ops1 -> row 2c seg0, ops2 -> row 2c+1 seg0), 4 per wave
#pragma unroll 1
  for (int i = 0; i < 4; ++i) {
    int r = wid * 4 + i;                         // 0..31
    int c = r >> 1;
    int oid = op_ids[(c0 + c) * 3 + 1 + (r & 1)];
    const float* src = op_emb + oid * DIM;
    _Float16* dst = Xin + (2 * c + (r & 1)) * SX;
#pragma unroll
    for (int p = 0; p < 8; ++p) { int k = lane + 32 * p; dst[k] = (_Float16)src[k]; }
  }
  __syncthreads();

  // ---- GEMM1: H = relu(Xin @ W1 + b1)  [32,768]x[768,512]
  // 32 n-tiles, 4 per wave; both m-tiles share each B fragment
#pragma unroll 1
  for (int i = 0; i < 4; ++i) {
    int nt = wid * 4 + i;                        // 0..31
    v8f acc0 = {}, acc1 = {};
    gemm_tile_pair(Xin, SX, P1, nt, 24, lane, acc0, acc1);
    store_tile_relu_f16(H, SH, 0, nt, acc0, b1, lane);
    store_tile_relu_f16(H, SH, 1, nt, acc1, b1, lane);
  }
  __syncthreads();

  // ---- GEMM2: Yf = H @ W2 + b2  [32,512]x[512,256]
  // 16 n-tiles, 2 per wave; both m-tiles share each B fragment
#pragma unroll 1
  for (int i = 0; i < 2; ++i) {
    int nt = wid * 2 + i;                        // 0..15
    v8f acc0 = {}, acc1 = {};
    gemm_tile_pair(H, SH, P2, nt, 16, lane, acc0, acc1);
    store_tile_f32(Yf, SYF, 0, nt, acc0, b2, lane);
    store_tile_f32(Yf, SYF, 1, nt, acc1, b2, lane);
  }
  __syncthreads();

  // ---- LN each of 32 rows -> L1h f16, 4 rows per wave
#pragma unroll 1
  for (int i = 0; i < 4; ++i) {
    int r = wid * 4 + i;
    ln_row_to_f16(Yf + r * SYF, L1h + r * DIM, gamma, beta, lane);
  }
  __syncthreads();

  // ---- Stage 2: build root_in into Xin rows 0..15
#pragma unroll 1
  for (int i = 0; i < 2; ++i) {
    int c = wid * 2 + i;                         // 0..15
    int oid = op_ids[(c0 + c) * 3 + 0];
    const float* src = op_emb + oid * DIM;
    _Float16* dst = Xin + c * SX;
#pragma unroll
    for (int p = 0; p < 8; ++p) {
      int k = lane + 32 * p;
      dst[k]           = (_Float16)src[k];
      dst[DIM + k]     = L1h[(2 * c) * DIM + k];
      dst[2 * DIM + k] = L1h[(2 * c + 1) * DIM + k];
    }
  }
  __syncthreads();

  // ---- GEMM1b: H[0:16] = relu(root_in @ W1 + b1), 32 tiles, 4/wave
#pragma unroll 1
  for (int i = 0; i < 4; ++i) {
    int nt = wid * 4 + i;
    v8f acc = gemm_tile(Xin, SX, 0, P1, nt, 24, lane);
    store_tile_relu_f16(H, SH, 0, nt, acc, b1, lane);
  }
  __syncthreads();

  // ---- GEMM2b: Yf[0:16] = H[0:16] @ W2 + b2, 16 tiles, 2/wave
#pragma unroll 1
  for (int i = 0; i < 2; ++i) {
    int nt = wid * 2 + i;
    v8f acc = gemm_tile(H, SH, 0, P2, nt, 16, lane);
    store_tile_f32(Yf, SYF, 0, nt, acc, b2, lane);
  }
  __syncthreads();

  // ---- LN root rows -> global table (f32), 2 rows per wave
#pragma unroll 1
  for (int i = 0; i < 2; ++i) {
    int c = wid * 2 + i;
    ln_row_to_f32(Yf + c * SYF, table + (size_t)(c0 + c) * DIM, gamma, beta, lane);
  }
}

// ---------------------------------------------------------------------------
// Gather: out[i,:] = (flat[i] <= 1) ? 0 : table[flat[i],:]
// ---------------------------------------------------------------------------
__global__ void gather_kernel(const int* __restrict__ flat,
                              const float* __restrict__ table,
                              float* __restrict__ out, int n) {
  int idx = blockIdx.x * blockDim.x + threadIdx.x;   // one float4 per thread
  if (idx >= n * 64) return;
  int i  = idx >> 6;
  int k4 = idx & 63;
  int ch = flat[i];
  v4f val = {};
  if (ch > 1) val = *(const v4f*)(table + (size_t)ch * DIM + k4 * 4);
  *(v4f*)(out + (size_t)i * DIM + k4 * 4) = val;
}

// ---------------------------------------------------------------------------
extern "C" void kernel_launch(void* const* d_in, const int* in_sizes, int n_in,
                              void* d_out, int out_size, void* d_ws, size_t ws_size,
                              hipStream_t stream) {
  const int*   struct_indices = (const int*)d_in[0];
  const int*   leaf_ids = (const int*)d_in[1];
  const int*   op_ids   = (const int*)d_in[2];
  const float* comp_emb = (const float*)d_in[3];
  const float* op_emb   = (const float*)d_in[4];
  const float* W1    = (const float*)d_in[5];
  const float* b1    = (const float*)d_in[6];
  const float* W2    = (const float*)d_in[7];
  const float* b2    = (const float*)d_in[8];
  const float* gamma = (const float*)d_in[9];
  const float* beta  = (const float*)d_in[10];

  char* ws = (char*)d_ws;
  float*    table = (float*)ws;     ws += (size_t)NCHARS * DIM * 4;   // 20,480,000 B
  _Float16* P1    = (_Float16*)ws;  ws += (size_t)DIM3 * DIM2 * 2;    // 786,432 B
  _Float16* P2    = (_Float16*)ws;                                    // 262,144 B

  // 1) pack weights into WMMA B-fragment order (f16)
  int packN = DIM3 * DIM2 + DIM2 * DIM;                // 524288
  pack_weights_kernel<<<(packN + 255) / 256, 256, 0, stream>>>(W1, W2, P1, P2);

  // 2) evaluate tree MLP per char -> root table
  hipFuncSetAttribute((const void*)tree_encode_kernel,
                      hipFuncAttributeMaxDynamicSharedMemorySize, SMEM_BYTES);
  tree_encode_kernel<<<NCHARS / CB, 256, SMEM_BYTES, stream>>>(
      leaf_ids, op_ids, comp_emb, op_emb, P1, P2, b1, b2, gamma, beta, table);

  // 3) gather to [B,S,D]
  int n = in_sizes[0];                                 // 32768 tokens
  int tot = n * 64;                                    // float4 units
  gather_kernel<<<(tot + 255) / 256, 256, 0, stream>>>(struct_indices, table,
                                                       (float*)d_out, n);
}
